// Multi_Head_Attention_29652454212583
// MI455X (gfx1250) — matrix operand order repaired and checked by reading, not compiled, not run on hardware
//
#include <hip/hip_runtime.h>
#include <math.h>

#define DD 512
#define NN 512
#define HH 8
#define BB 8
#define NEGC 3e31f

typedef __attribute__((ext_vector_type(16))) __bf16 v16bf;
typedef __attribute__((ext_vector_type(8)))  float v8f;
typedef __attribute__((ext_vector_type(4)))  int   v4i;

#if __has_builtin(__builtin_amdgcn_global_load_async_to_lds_b128)
  #define HAS_ASYNC 1
#else
  #define HAS_ASYNC 0
#endif

#if HAS_ASYNC
  #if __has_builtin(__builtin_amdgcn_s_wait_asynccnt)
    #define WAIT_ASYNC(n) __builtin_amdgcn_s_wait_asynccnt(n)
  #else
    #define WAIT_ASYNC(n) asm volatile("s_wait_asynccnt " #n ::: "memory")
  #endif
#else
  #define WAIT_ASYNC(n) ((void)0)
#endif

__device__ __forceinline__ unsigned short f2bf(float f) {
  unsigned u = __float_as_uint(f);
  u += 0x7fffu + ((u >> 16) & 1u);          // round-to-nearest-even
  return (unsigned short)(u >> 16);
}

// 16B-per-lane async copy global -> LDS (ASYNCcnt) with sync fallback.
__device__ __forceinline__ void async_copy_b128(const unsigned short* g,
                                                unsigned short* l) {
#if HAS_ASYNC
  typedef __attribute__((address_space(1))) v4i* gp_t;  // global int4*
  typedef __attribute__((address_space(3))) v4i* lp_t;  // LDS int4*
  __builtin_amdgcn_global_load_async_to_lds_b128(
      (gp_t)(v4i*)g, (lp_t)(v4i*)l, 0, 0);
#else
  *reinterpret_cast<uint4*>(l) = *reinterpret_cast<const uint4*>(g);
#endif
}

// ---- fragment loads, ISA 16-bit WMMA VGPR layouts (CDNA5 ISA §7.12.2) ----
// A stored [M][K] K-contiguous: lane<16 K={0..7,16..23}, lane>=16 +8.
__device__ __forceinline__ v16bf frag_a(const unsigned short* p0, int lda,
                                        int m0, int k0, int lane) {
  int lh = lane >> 4, m = lane & 15;
  const unsigned short* p = p0 + (size_t)(m0 + m) * lda + k0 + lh * 8;
  union { uint4 q[2]; v16bf v; } u;
  u.q[0] = *reinterpret_cast<const uint4*>(p);
  u.q[1] = *reinterpret_cast<const uint4*>(p + 16);
  return u.v;
}
// B stored [N][K] K-contiguous: lane<16 K={0..7,16..23}, lane>=16 +8.
__device__ __forceinline__ v16bf frag_b(const unsigned short* p0, int ldb,
                                        int n0, int k0, int lane) {
  int lh = lane >> 4, n = lane & 15;
  const unsigned short* p = p0 + (size_t)(n0 + n) * ldb + k0 + lh * 8;
  union { uint4 q[2]; v16bf v; } u;
  u.q[0] = *reinterpret_cast<const uint4*>(p);
  u.q[1] = *reinterpret_cast<const uint4*>(p + 16);
  return u.v;
}

__device__ __forceinline__ v8f wmma_bf16(v16bf a, v16bf b, v8f c) {
  return __builtin_amdgcn_wmma_f32_16x16x32_bf16(false, a, false, b,
                                                 (short)0, c, false, false);
}

// ---------------------------------------------------------------- utilities

__global__ __launch_bounds__(256) void transpose_cvt_x(
    const float* __restrict__ x, unsigned short* __restrict__ xt) {
  __shared__ float tile[32][33];
  int b = blockIdx.z;
  int n0 = blockIdx.x * 32, i0 = blockIdx.y * 32;
  const float* xb = x + (size_t)b * DD * NN;
  unsigned short* xtb = xt + (size_t)b * DD * NN;
  int c = threadIdx.x & 31, r0 = threadIdx.x >> 5;
  for (int p = 0; p < 4; p++) {
    int r = r0 + p * 8;
    tile[r][c] = xb[(size_t)(i0 + r) * NN + n0 + c];
  }
  __syncthreads();
  for (int p = 0; p < 4; p++) {
    int r = r0 + p * 8;
    xtb[(size_t)(n0 + r) * DD + i0 + c] = f2bf(tile[c][r]);
  }
}

__global__ __launch_bounds__(256) void cvt_bf16(
    const float* __restrict__ src, unsigned short* __restrict__ dst, int n4) {
  int idx = blockIdx.x * 256 + threadIdx.x;
  if (idx >= n4) return;
  float4 v = reinterpret_cast<const float4*>(src)[idx];
  union { unsigned short s[4]; uint2 q; } u;
  u.s[0] = f2bf(v.x); u.s[1] = f2bf(v.y);
  u.s[2] = f2bf(v.z); u.s[3] = f2bf(v.w);
  reinterpret_cast<uint2*>(dst)[idx] = u.q;
}

__global__ __launch_bounds__(256) void pad_kernel(
    const float* __restrict__ x, float* __restrict__ padNeg) {
  int b = blockIdx.y;
  int j = blockIdx.x * 256 + threadIdx.x;
  const float* xb = x + (size_t)b * DD * NN;
  bool ok = true;
  for (int i = 0; i < DD; i++) ok = ok && (xb[(size_t)i * NN + j] != 0.f);
  padNeg[b * NN + j] = ok ? 0.f : -NEGC;
}

// ------------------------------------------------------------- generic GEMM
// C(512,512) = alpha * A(M,K) * B(N,K)^T, bf16, f32 accumulate.
// 128x128 block tile, 8 waves (2m x 4n), wave tile 64x32 (4x2 frags).
// A/B tiles double-buffered in LDS via GLOBAL_LOAD_ASYNC_TO_LDS_B128;
// LDS rows padded to 40 elems (80 B) for conflict-free b128 reads.
// cmode: 0 = f32 [M][N], 1 = bf16 [M][N], 2 = bf16 transposed [N][M].
#define LP 40  // padded LDS row stride (elements)

__global__ __launch_bounds__(256) void gemm_bf16_kernel(
    const unsigned short* __restrict__ A, long long sAy, long long sAz,
    const unsigned short* __restrict__ Bm, long long sBy, long long sBz,
    void* __restrict__ C, long long sCy, long long sCz,
    int K, int lda, int ldb, int ldc, int cmode, float alpha) {
  __shared__ __align__(16) unsigned short Ab[2][128][LP];
  __shared__ __align__(16) unsigned short Bb[2][128][LP];

  A  += (size_t)blockIdx.y * sAy + (size_t)blockIdx.z * sAz;
  Bm += (size_t)blockIdx.y * sBy + (size_t)blockIdx.z * sBz;
  size_t coff = (size_t)blockIdx.y * sCy + (size_t)blockIdx.z * sCz;

  int tid = threadIdx.x, lane = tid & 31, wid = tid >> 5;
  int m0b = (blockIdx.x & 3) * 128;
  int n0b = (blockIdx.x >> 2) * 128;
  int mw = (wid & 1) * 64;    // wave m-offset inside tile
  int nw = (wid >> 1) * 32;   // wave n-offset inside tile

  auto stage = [&](int buf, int k0) {
    for (int t = 0; t < 2; t++) {
      int c = tid + t * 256;              // 0..511  (row, 16B-quad)
      int row = c >> 2, kq = (c & 3) * 8; // 4 quads of 8 bf16 = 32 k-elems
      async_copy_b128(A + (size_t)(m0b + row) * lda + k0 + kq,
                      &Ab[buf][row][kq]);
      async_copy_b128(Bm + (size_t)(n0b + row) * ldb + k0 + kq,
                      &Bb[buf][row][kq]);
    }
  };

  v8f z; for (int r = 0; r < 8; r++) z[r] = 0.f;
  v8f acc[4][2];
  for (int fm = 0; fm < 4; fm++)
    for (int fn = 0; fn < 2; fn++) acc[fm][fn] = z;

  int nsteps = K >> 5;
  stage(0, 0);
  for (int s = 0; s < nsteps; s++) {
    int cur = s & 1;
    if (s + 1 < nsteps) {
      stage(cur ^ 1, (s + 1) * 32);
      WAIT_ASYNC(4);                 // 4 newly-issued ops may stay in flight
    } else {
      WAIT_ASYNC(0);
    }
    __syncthreads();                 // staged tile visible to all waves

    const unsigned short* Abase = &Ab[cur][0][0];
    const unsigned short* Bbase = &Bb[cur][0][0];
    v16bf af[4], bf[2];
#pragma unroll
    for (int fm = 0; fm < 4; fm++) af[fm] = frag_a(Abase, LP, mw + fm * 16, 0, lane);
#pragma unroll
    for (int fn = 0; fn < 2; fn++) bf[fn] = frag_b(Bbase, LP, nw + fn * 16, 0, lane);
#pragma unroll
    for (int fm = 0; fm < 4; fm++)
#pragma unroll
      for (int fn = 0; fn < 2; fn++)
        acc[fm][fn] = wmma_bf16(af[fm], bf[fn], acc[fm][fn]);

    __syncthreads();                 // all reads done before buffer reuse
  }

  int lh = lane >> 4, ln = lane & 15;
  if (cmode == 2) {
    unsigned short* Cc = (unsigned short*)C + coff;
    for (int fm = 0; fm < 4; fm++)
      for (int fn = 0; fn < 2; fn++) {
        union { unsigned short s[8]; uint4 q; } u;
        for (int r = 0; r < 8; r++) u.s[r] = f2bf(acc[fm][fn][r] * alpha);
        size_t n = (size_t)(n0b + nw + fn * 16 + ln);
        *reinterpret_cast<uint4*>(Cc + n * ldc + m0b + mw + fm * 16 + lh * 8) = u.q;
      }
  } else if (cmode == 1) {
    unsigned short* Cc = (unsigned short*)C + coff;
    for (int fm = 0; fm < 4; fm++)
      for (int fn = 0; fn < 2; fn++)
        for (int r = 0; r < 8; r++)
          Cc[(size_t)(m0b + mw + fm * 16 + r + 8 * lh) * ldc +
             n0b + nw + fn * 16 + ln] = f2bf(acc[fm][fn][r] * alpha);
  } else {
    float* Cf = (float*)C + coff;
    for (int fm = 0; fm < 4; fm++)
      for (int fn = 0; fn < 2; fn++)
        for (int r = 0; r < 8; r++)
          Cf[(size_t)(m0b + mw + fm * 16 + r + 8 * lh) * ldc +
             n0b + nw + fn * 16 + ln] = acc[fm][fn][r] * alpha;
  }
}

// ------------------------------------------------- fused scores + softmax
__global__ __launch_bounds__(256) void attn_kernel(
    const unsigned short* __restrict__ Qt, const unsigned short* __restrict__ Kt,
    const float* __restrict__ padNeg, unsigned short* __restrict__ At) {
  __shared__ float S[16][512];
  int b = blockIdx.z, h = blockIdx.y, i0 = blockIdx.x * 16;
  size_t bh = (size_t)(b * HH + h) * NN * DD;
  const unsigned short* Q  = Qt + bh;
  const unsigned short* Km = Kt + bh;
  int tid = threadIdx.x, lane = tid & 31, wid = tid >> 5;
  int jw = wid * 64;

  v8f z; for (int r = 0; r < 8; r++) z[r] = 0.f;
  v8f acc[4] = {z, z, z, z};
  for (int k0 = 0; k0 < DD; k0 += 32) {
    v16bf a = frag_a(Q, DD, i0, k0, lane);
    for (int f = 0; f < 4; f++) {
      v16bf bb = frag_b(Km, DD, jw + f * 16, k0, lane);
      acc[f] = wmma_bf16(a, bb, acc[f]);
    }
  }
  int lh = lane >> 4, ln = lane & 15;
  for (int f = 0; f < 4; f++) {
    int j = jw + f * 16 + ln;
    float pn = padNeg[b * NN + j];
    for (int r = 0; r < 8; r++) {
      int i = i0 + r + 8 * lh;
      S[r + 8 * lh][j] = acc[f][r] + pn + ((j <= i) ? 0.f : -NEGC);
    }
  }
  __syncthreads();
  for (int rr = 0; rr < 2; rr++) {     // each wave owns 2 rows
    int row = wid * 2 + rr;
    int i = i0 + row;
    float mx = -INFINITY;
    for (int c = lane; c < NN; c += 32) mx = fmaxf(mx, S[row][c]);
    for (int off = 16; off > 0; off >>= 1)
      mx = fmaxf(mx, __shfl_xor(mx, off, 32));
    float sum = 0.f;
    for (int c = lane; c < NN; c += 32) {
      float p = __expf(S[row][c] - mx);
      S[row][c] = p;
      sum += p;
    }
    for (int off = 16; off > 0; off >>= 1) sum += __shfl_xor(sum, off, 32);
    float inv = 1.f / sum;
    for (int c = lane; c < NN; c += 32)
      At[bh + (size_t)c * NN + i] = f2bf(S[row][c] * inv);
  }
}

// --------------------------------------------------------------- launcher
extern "C" void kernel_launch(void* const* d_in, const int* in_sizes, int n_in,
                              void* d_out, int out_size, void* d_ws,
                              size_t ws_size, hipStream_t stream) {
  const float* x  = (const float*)d_in[0];
  const float* Uq = (const float*)d_in[1];
  const float* Uk = (const float*)d_in[2];
  const float* Uv = (const float*)d_in[3];
  const float* W  = (const float*)d_in[4];
  float* out = (float*)d_out;

  char* ws = (char*)d_ws;
  size_t off = 0;
  auto alloc = [&](size_t bytes) -> void* {
    void* p = ws + off;
    off += (bytes + 255) & ~(size_t)255;
    return p;
  };
  const size_t E = (size_t)DD * NN;            // 262144 per matrix
  unsigned short* xt  = (unsigned short*)alloc(BB * E * 2);
  unsigned short* Uqb = (unsigned short*)alloc(HH * E * 2);
  unsigned short* Ukb = (unsigned short*)alloc(HH * E * 2);
  unsigned short* Uvb = (unsigned short*)alloc(HH * E * 2);
  unsigned short* Wb  = (unsigned short*)alloc((size_t)DD * HH * DD * 2);
  float* padNeg       = (float*)alloc((size_t)BB * NN * 4);
  unsigned short* Qt  = (unsigned short*)alloc((size_t)BB * HH * E * 2);
  unsigned short* Kt  = (unsigned short*)alloc((size_t)BB * HH * E * 2);
  unsigned short* Vn  = (unsigned short*)alloc((size_t)BB * HH * E * 2);
  unsigned short* At  = (unsigned short*)alloc((size_t)BB * HH * E * 2);
  unsigned short* Yt  = Qt;  // Qt is dead after attn_kernel; reuse for Yt

  const long long sE  = (long long)E;          // 262144
  const long long sBH = (long long)HH * E;     // 2097152

  transpose_cvt_x<<<dim3(16, 16, BB), 256, 0, stream>>>(x, xt);
  cvt_bf16<<<2048, 256, 0, stream>>>(Uq, Uqb, (int)(HH * E / 4));
  cvt_bf16<<<2048, 256, 0, stream>>>(Uk, Ukb, (int)(HH * E / 4));
  cvt_bf16<<<2048, 256, 0, stream>>>(Uv, Uvb, (int)(HH * E / 4));
  cvt_bf16<<<2048, 256, 0, stream>>>(W, Wb, (int)((size_t)DD * HH * DD / 4));
  pad_kernel<<<dim3(2, BB), 256, 0, stream>>>(x, padNeg);

  // Q (scaled by 1/8, stored transposed [n][d]):  y=b, z=h
  gemm_bf16_kernel<<<dim3(16, BB, HH), 256, 0, stream>>>(
      Uqb, 0, sE, xt, sE, 0, Qt, sBH, sE,
      512, 512, 512, 512, 2, 0.125f);
  // K (stored transposed [n][d])
  gemm_bf16_kernel<<<dim3(16, BB, HH), 256, 0, stream>>>(
      Ukb, 0, sE, xt, sE, 0, Kt, sBH, sE,
      512, 512, 512, 512, 2, 1.0f);
  // V (stored natural [d][n])
  gemm_bf16_kernel<<<dim3(16, BB, HH), 256, 0, stream>>>(
      Uvb, 0, sE, xt, sE, 0, Vn, sBH, sE,
      512, 512, 512, 512, 1, 1.0f);

  // scores + mask + softmax -> At[j][i]
  attn_kernel<<<dim3(32, HH, BB), 256, 0, stream>>>(Qt, Kt, padNeg, At);

  // Y[d][j] = V[d][:] * A[:][j]; stored transposed into Yt[b][j][h*512+d]
  gemm_bf16_kernel<<<dim3(16, BB, HH), 256, 0, stream>>>(
      Vn, sBH, sE, At, sBH, sE, Yt, sBH, 512,
      512, 512, 512, 4096, 2, 1.0f);

  // out[b][e][n] = W[e][:] * Yt[b][n][:]^T   (K = 4096), f32 output
  gemm_bf16_kernel<<<dim3(16, BB, 1), 256, 0, stream>>>(
      Wb, 0, 0, Yt, sBH, 0, out, sE, 0,
      4096, 4096, 4096, 512, 0, 1.0f);
}
